// SparseResBlockC2S3d_14568529068654
// MI455X (gfx1250) — matrix operand order repaired and checked by reading, not compiled, not run on hardware
//
#include <hip/hip_runtime.h>

// ---------------------------------------------------------------------------
// SparseResBlock (32^3 grid, C=64) for MI455X / gfx1250 (wave32, WMMA).
// GEMMs use v_wmma_f32_16x16x32_bf16. A operands staged via double-buffered
// LDS tiles (ds_load_b128 in a contiguous-k A layout) with software-pipelined
// gathers. B operands pre-packed to the per-lane WMMA B layout (two
// contiguous global_load_b128 each) and reused across 2 (conv1) / 4 (conv2)
// M row-groups per wave to cut L2 weight traffic, which is the bandwidth
// limiter at these tile sizes. conv1's LN epilogue is done in-registers via
// 16-lane shfl_xor butterflies (each wave owns a full 64-col LN group).
// ---------------------------------------------------------------------------

typedef __bf16  bf16_t;
typedef bf16_t  v16bf __attribute__((ext_vector_type(16)));
typedef float   v8f   __attribute__((ext_vector_type(8)));
typedef unsigned short u16;

#define NVOX   32768     // D^3
#define CCH    64        // channels
#define OUT1   512       // 8*C
#define NROW2  262144    // 8*NVOX
#define TAPS   27
#define EPS_F  1e-6f

union Op16 { v16bf v; uint4 q[2]; };

__device__ __forceinline__ u16 f2bf(float f) {
    union { bf16_t h; u16 u; } c;
    c.h = (bf16_t)f;          // RNE f32 -> bf16
    return c.u;
}

__device__ __forceinline__ float silu_f(float y) {
    return y / (1.0f + __expf(-y));
}

__device__ __forceinline__ v8f wmma_bf16(v16bf a, v16bf b, v8f c) {
    return __builtin_amdgcn_wmma_f32_16x16x32_bf16(
        false, a, false, b, (short)0, c, false, false);
}

// Branch-free gather: clamp invalid index to 0, zero-select the payload.
__device__ __forceinline__ uint4 gather16B(const u16* __restrict__ src,
                                           int idx, int colh) {
    const u16* p = src + (size_t)(idx < 0 ? 0 : idx) * CCH + colh;
    uint4 d = *(const uint4*)p;
    if (idx < 0) { d.x = 0u; d.y = 0u; d.z = 0u; d.w = 0u; }
    return d;
}

// ---------------------------------------------------------------------------
// Pack W (TAPS, CCH, outW) f32 -> bf16 in WMMA B-operand layout:
//   packed[((tap*nTiles + tile)*2 + kc)*512 + lane*16 + e]
//     = W[tap][kc*32 + (lane>=16 ? 16:0) + e][tile*16 + (lane&15)]
// ---------------------------------------------------------------------------
__global__ void pack_w_kernel(const float* __restrict__ W, u16* __restrict__ P,
                              int nTiles, int outW) {
    int t    = blockIdx.x * blockDim.x + threadIdx.x;
    int lane = t & 31;
    int kc   = (t >> 5) & 1;
    int tt   = t >> 6;               // tap*nTiles + tile
    int tile = tt % nTiles;
    int tap  = tt / nTiles;
    if (tap >= TAPS) return;
    int kbase = kc * 32 + ((lane & 16) ? 16 : 0);
    int col   = tile * 16 + (lane & 15);
    const float* w = W + (size_t)tap * CCH * outW;
    u16* dst = P + (size_t)t * 16;
#pragma unroll
    for (int e = 0; e < 16; ++e)
        dst[e] = f2bf(w[(size_t)(kbase + e) * outW + col]);
}

// ---------------------------------------------------------------------------
// Per-voxel: subdiv = x @ W_sub + b_sub (output tuple elem 1 + keep mask),
// and h0 = bf16( silu( layernorm(x, g1, b1) ) ).
// ---------------------------------------------------------------------------
__global__ void subdiv_ln_kernel(const float* __restrict__ x,
                                 const float* __restrict__ g1,
                                 const float* __restrict__ b1,
                                 const float* __restrict__ Wsub,
                                 const float* __restrict__ bsub,
                                 float* __restrict__ subdiv_out,
                                 unsigned char* __restrict__ keep,
                                 u16* __restrict__ h0) {
    __shared__ float sW[CCH * 8];
    __shared__ float sG[CCH], sB[CCH], sBs[8];
    int tid = threadIdx.x;
    for (int i = tid; i < CCH * 8; i += blockDim.x) sW[i] = Wsub[i];
    for (int i = tid; i < CCH; i += blockDim.x) { sG[i] = g1[i]; sB[i] = b1[i]; }
    if (tid < 8) sBs[tid] = bsub[tid];
    __syncthreads();

    int r = blockIdx.x * blockDim.x + tid;
    if (r >= NVOX) return;

    const float* xr = x + (size_t)r * CCH;
    float f[CCH];
    float s = 0.f, s2 = 0.f;
#pragma unroll 8
    for (int i = 0; i < CCH; ++i) { f[i] = xr[i]; s += f[i]; s2 += f[i] * f[i]; }

    float acc[8];
#pragma unroll
    for (int k = 0; k < 8; ++k) acc[k] = sBs[k];
#pragma unroll 4
    for (int i = 0; i < CCH; ++i)
#pragma unroll
        for (int k = 0; k < 8; ++k) acc[k] += f[i] * sW[i * 8 + k];
#pragma unroll
    for (int k = 0; k < 8; ++k) {
        subdiv_out[(size_t)r * 8 + k] = acc[k];
        keep[(size_t)r * 8 + k] = (acc[k] > 0.f) ? 1 : 0;
    }

    float mean = s * (1.f / CCH);
    float var  = s2 * (1.f / CCH) - mean * mean;
    float rinv = rsqrtf(var + EPS_F);
    u16* o = h0 + (size_t)r * CCH;
#pragma unroll 8
    for (int i = 0; i < CCH; ++i) {
        float y = (f[i] - mean) * rinv * sG[i] + sB[i];
        o[i] = f2bf(silu_f(y));
    }
}

// ---------------------------------------------------------------------------
// conv1: 32 voxels x 512 outputs per block (8 waves; each wave: 4 col-tiles
// x 2 row-groups = 16 WMMA/tap, B operands reused across row-groups).
// Double-buffered LDS A tile, software-pipelined gather, 1 barrier/tap.
// Epilogue: each wave owns a complete 64-col LN group per reshaped row
// (g = voxel*8 + wave), so bias + keep mask + LN + SiLU run entirely in
// registers with 16-lane shfl_xor reductions, writing bf16 h1 directly.
// ---------------------------------------------------------------------------
__global__ void __launch_bounds__(256)
conv1_kernel(const u16* __restrict__ h0, const int* __restrict__ nbr1,
             const u16* __restrict__ pW1, const float* __restrict__ bc1,
             const unsigned char* __restrict__ keep, u16* __restrict__ h1) {
    __shared__ __align__(16) u16 sA[2][32][72];   // double-buffered A tiles
    __shared__ int sNbr[32][28];                  // block's neighbor table

    int tid  = threadIdx.x;
    int lane = tid & 31;
    int wave = tid >> 5;                // 0..7 -> 64-col slice
    int rb   = blockIdx.x * 32;         // voxel base (32 voxels)

    for (int i = tid; i < 32 * TAPS; i += 256)
        sNbr[i / TAPS][i % TAPS] = nbr1[(size_t)rb * TAPS + i];
    __syncthreads();

    v8f acc[2][4] = {};
    int grow   = tid >> 3;              // gather row 0..31
    int gcol   = (tid & 7) * 8;         // 8 halves per thread
    int lrow   = lane & 15;             // A-matrix M row for this lane
    int koffA0 = (lane & 16) ? 16 : 0;  // contiguous-k A layout half-offset

    {   // prologue gather: tap 0 -> buffer 0
        uint4 d0 = gather16B(h0, sNbr[grow][0], gcol);
        *(uint4*)&sA[0][grow][gcol] = d0;
    }
    __syncthreads();

    const u16* bwave = pW1 + (size_t)lane * 16 + (size_t)wave * (4 * 2 * 512);

    for (int tap = 0; tap < TAPS; ++tap) {
        int  cur  = tap & 1;
        bool more = (tap + 1 < TAPS);
        uint4 dn;
        if (more) dn = gather16B(h0, sNbr[grow][tap + 1], gcol);  // issue early

        const u16* bbase = bwave + (size_t)tap * (32 * 2 * 512);
        if (more) __builtin_prefetch(bbase + 32 * 2 * 512, 0, 3);
#pragma unroll
        for (int kc = 0; kc < 2; ++kc) {
            int kh = kc * 32 + koffA0;
            Op16 b[4];
#pragma unroll
            for (int ct = 0; ct < 4; ++ct) {
                const u16* bp = bbase + (size_t)(ct * 2 + kc) * 512;
                b[ct].q[0] = *(const uint4*)bp;
                b[ct].q[1] = *(const uint4*)(bp + 8);
            }
#pragma unroll
            for (int rg = 0; rg < 2; ++rg) {       // B reused across row-groups
                Op16 a;
                a.q[0] = *(const uint4*)&sA[cur][rg * 16 + lrow][kh];
                a.q[1] = *(const uint4*)&sA[cur][rg * 16 + lrow][kh + 8];
#pragma unroll
                for (int ct = 0; ct < 4; ++ct)
                    acc[rg][ct] = wmma_bf16(a.v, b[ct].v, acc[rg][ct]);
            }
        }
        if (more) *(uint4*)&sA[1 - cur][grow][gcol] = dn;     // commit next tap
        __syncthreads();
    }

    // In-register epilogue. D layout: acc[rg][ct][r] = element
    //   (M = r + 8*(lane>=16), col = wave*64 + ct*16 + (lane&15)).
    // LN group = 64 cols = reduce over ct and the owning 16-lane half.
    float bs[4];
#pragma unroll
    for (int ct = 0; ct < 4; ++ct)
        bs[ct] = bc1[wave * 64 + ct * 16 + (lane & 15)];
#pragma unroll
    for (int rg = 0; rg < 2; ++rg) {
#pragma unroll
        for (int r = 0; r < 8; ++r) {
            float e0 = acc[rg][0][r] + bs[0];
            float e1 = acc[rg][1][r] + bs[1];
            float e2 = acc[rg][2][r] + bs[2];
            float e3 = acc[rg][3][r] + bs[3];
            float s  = e0 + e1 + e2 + e3;
            float s2 = e0 * e0 + e1 * e1 + e2 * e2 + e3 * e3;
#pragma unroll
            for (int m = 1; m < 16; m <<= 1) {     // stays within half-wave
                s  += __shfl_xor(s,  m, 32);
                s2 += __shfl_xor(s2, m, 32);
            }
            float mean = s * (1.f / CCH);
            float var  = s2 * (1.f / CCH) - mean * mean;
            float rinv = rsqrtf(var + EPS_F);
            int M = r + ((lane & 16) ? 8 : 0);
            int g = (rb + rg * 16 + M) * 8 + wave;   // row in (8N,64)
            float kp = keep[g] ? 1.f : 0.f;          // masked row -> 0 (== LN(0))
            u16* o = h1 + (size_t)g * CCH + (lane & 15);
            o[0]  = f2bf(silu_f((e0 - mean) * rinv) * kp);
            o[16] = f2bf(silu_f((e1 - mean) * rinv) * kp);
            o[32] = f2bf(silu_f((e2 - mean) * rinv) * kp);
            o[48] = f2bf(silu_f((e3 - mean) * rinv) * kp);
        }
    }
}

// ---------------------------------------------------------------------------
// conv2: 64 rows x 64 outputs per block (4 waves; each wave: one col-tile
// x 4 row-groups = 8 WMMA/tap, each B operand reused 4x).
// Epilogue: bias, keep mask, skip-add (repeat(xs,8)), f32 store to d_out.
// ---------------------------------------------------------------------------
__global__ void __launch_bounds__(128)
conv2_kernel(const u16* __restrict__ h1, const int* __restrict__ nbr2,
             const u16* __restrict__ pW2, const float* __restrict__ bc2,
             const unsigned char* __restrict__ keep, const float* __restrict__ x,
             float* __restrict__ out) {
    __shared__ __align__(16) u16 sA[2][64][72];
    __shared__ int sNbr[64][28];

    int tid  = threadIdx.x;
    int lane = tid & 31;
    int wave = tid >> 5;                 // 0..3 = col tile
    int rb   = blockIdx.x * 64;          // row base in (8N) space

    for (int i = tid; i < 64 * TAPS; i += 128)
        sNbr[i / TAPS][i % TAPS] = nbr2[(size_t)rb * TAPS + i];
    __syncthreads();

    v8f acc[4] = {};
    int grow   = tid >> 1;               // gather row 0..63
    int gcol   = (tid & 1) * 32;         // 16 halves (two b128) per thread
    int lrow   = lane & 15;
    int koffA0 = (lane & 16) ? 16 : 0;

    {
        int idx = sNbr[grow][0];
        uint4 d0 = gather16B(h1, idx, gcol);
        uint4 d1 = gather16B(h1, idx, gcol + 8);
        *(uint4*)&sA[0][grow][gcol]     = d0;
        *(uint4*)&sA[0][grow][gcol + 8] = d1;
    }
    __syncthreads();

    const u16* bwave = pW2 + (size_t)lane * 16 + (size_t)wave * (2 * 512);

    for (int tap = 0; tap < TAPS; ++tap) {
        int  cur  = tap & 1;
        bool more = (tap + 1 < TAPS);
        uint4 dna, dnb;
        if (more) {
            int idx = sNbr[grow][tap + 1];
            dna = gather16B(h1, idx, gcol);
            dnb = gather16B(h1, idx, gcol + 8);
        }

        const u16* bbase = bwave + (size_t)tap * (4 * 2 * 512);
        if (more) __builtin_prefetch(bbase + 4 * 2 * 512, 0, 3);
#pragma unroll
        for (int kc = 0; kc < 2; ++kc) {
            int kh = kc * 32 + koffA0;
            const u16* bp = bbase + (size_t)kc * 512;
            Op16 b;
            b.q[0] = *(const uint4*)bp;
            b.q[1] = *(const uint4*)(bp + 8);
#pragma unroll
            for (int rg = 0; rg < 4; ++rg) {       // B reused 4x
                Op16 a;
                a.q[0] = *(const uint4*)&sA[cur][rg * 16 + lrow][kh];
                a.q[1] = *(const uint4*)&sA[cur][rg * 16 + lrow][kh + 8];
                acc[rg] = wmma_bf16(a.v, b.v, acc[rg]);
            }
        }
        if (more) {
            *(uint4*)&sA[1 - cur][grow][gcol]     = dna;
            *(uint4*)&sA[1 - cur][grow][gcol + 8] = dnb;
        }
        __syncthreads();
    }

    int col  = wave * 16 + (lane & 15);
    float bsv = bc2[col];
    int xcol = col >> 3;                 // repeat(xs, 8, axis=1)
#pragma unroll
    for (int rg = 0; rg < 4; ++rg) {
#pragma unroll
        for (int r = 0; r < 8; ++r) {
            int M = r + ((lane & 16) ? 8 : 0);
            int g = rb + rg * 16 + M;
            float kp   = keep[g] ? 1.f : 0.f;
            float skip = kp * x[(size_t)g * 8 + xcol];   // xs already masked
            out[(size_t)g * CCH + col] = (acc[rg][r] + bsv) * kp + skip;
        }
    }
}

// ---------------------------------------------------------------------------
extern "C" void kernel_launch(void* const* d_in, const int* in_sizes, int n_in,
                              void* d_out, int out_size, void* d_ws, size_t ws_size,
                              hipStream_t stream) {
    (void)in_sizes; (void)n_in; (void)out_size; (void)ws_size;

    const float* x    = (const float*)d_in[0];
    const float* g1   = (const float*)d_in[1];
    const float* b1   = (const float*)d_in[2];
    const float* Wsub = (const float*)d_in[3];
    const float* bsub = (const float*)d_in[4];
    const float* W1   = (const float*)d_in[5];
    const float* bc1  = (const float*)d_in[6];
    const float* W2   = (const float*)d_in[7];
    const float* bc2  = (const float*)d_in[8];
    const int*   nbr1 = (const int*)d_in[9];
    const int*   nbr2 = (const int*)d_in[10];

    float* out_main = (float*)d_out;                        // (8N, 64)
    float* out_sub  = (float*)d_out + (size_t)NROW2 * CCH;  // (N, 8)

    char*  ws  = (char*)d_ws;
    size_t off = 0;
    auto take = [&](size_t bytes) -> char* {
        char* p = ws + off;
        off += (bytes + 255) & ~(size_t)255;
        return p;
    };
    u16* h0            = (u16*)take((size_t)NVOX  * CCH * 2);           //  4 MB
    u16* h1            = (u16*)take((size_t)NROW2 * CCH * 2);           // 32 MB
    unsigned char* kp  = (unsigned char*)take((size_t)NROW2);           // .25 MB
    u16* pW1           = (u16*)take((size_t)TAPS * 32 * 2 * 512 * 2);   // 1.7 MB
    u16* pW2           = (u16*)take((size_t)TAPS * 4  * 2 * 512 * 2);   // .21 MB

    pack_w_kernel<<<(TAPS * 32 * 64) / 256, 256, 0, stream>>>(W1, pW1, 32, OUT1);
    pack_w_kernel<<<(TAPS * 4  * 64) / 256, 256, 0, stream>>>(W2, pW2, 4, CCH);
    subdiv_ln_kernel<<<NVOX / 256, 256, 0, stream>>>(x, g1, b1, Wsub, bsub,
                                                     out_sub, kp, h0);
    conv1_kernel<<<NVOX / 32, 256, 0, stream>>>(h0, nbr1, pW1, bc1, kp, h1);
    conv2_kernel<<<NROW2 / 64, 128, 0, stream>>>(h1, nbr2, pW2, bc2, kp, x, out_main);
}
